// ModelConstruct_53180285059114
// MI455X (gfx1250) — compile-verified, matrix operand order checked by reading, not run on hardware
//
#include <hip/hip_runtime.h>
#include <hip/hip_bf16.h>

// Problem dimensions (fixed by the reference).
#define B_     128
#define T_     1024
#define D_     256
#define UNITS_ 128
#define GATES_ 512   // 4*UNITS
#define HID_   64
#define OUT_   8

typedef __attribute__((ext_vector_type(16))) __bf16 v16bf;
typedef __attribute__((ext_vector_type(8)))  float  v8f;

// Workspace layout (dword offsets).
#define WSW_OFF    0u        // W  frags: [kk=8][nt=32][lane=32][v=8] -> 65536 dw (256 KB)
#define WSU_OFF    65536u    // U  frags: [kk=4][nt=32][lane=32][v=8] -> 32768 dw (128 KB)
#define WSWD_OFF   98304u    // Wd frags: [kk=4][nt=8 ][lane=32][v=8] ->  8192 dw ( 32 KB)
#define HSEL_OFF   106496u   // gathered hidden states [B][UNITS] f32 -> 16384 dw ( 64 KB)
#define GT_OFF     122880u   // time decay, transposed: gT[T][B] f32   -> 131072 dw (512 KB)
#define ZFRAG_OFF  253952u   // Z=x@W bf16 C/D-frags [mt=8][t=1024][nt=32][lane=32][w=4]
#define ZFRAG_DW   33554432u //   -> 128 MB
#define ZSTEP_DW   4096u     // dwords per (mt, t): 32 tiles * 32 lanes * 4

// ---------- helpers ----------
__device__ __forceinline__ unsigned f2bf1(float f) {
    unsigned u = __float_as_uint(f);
    return (u + 0x7FFFu + ((u >> 16) & 1u)) >> 16;   // RNE f32 -> bf16 (low 16 bits)
}
__device__ __forceinline__ unsigned packbf(float lo, float hi) {
    return (f2bf1(lo) & 0xFFFFu) | ((f2bf1(hi) & 0xFFFFu) << 16);
}
// A/B fragment forward mapping: (lane, dword v) -> k-within-32 (ISA 16-bit layout)
__device__ __forceinline__ unsigned frag_kin(unsigned lane, unsigned v) {
    return ((v & 3u) * 2u) + ((v >> 2) * 16u) + ((lane >> 4) * 8u);
}
__device__ __forceinline__ float sigm(float x) {
    return __fdividef(1.0f, 1.0f + __expf(-x));
}
__device__ __forceinline__ float tanh_fast(float x) {
    float e = __expf(2.0f * x);
    return 1.0f - __fdividef(2.0f, e + 1.0f);
}
__device__ __forceinline__ v8f wmma_bf16(v16bf a, v16bf b, v8f c) {
    return __builtin_amdgcn_wmma_f32_16x16x32_bf16(false, a, false, b, (short)0, c, false, false);
}
__device__ __forceinline__ uint4 pack_acc(v8f a) {
    uint4 o;
    o.x = packbf(a[0], a[1]); o.y = packbf(a[2], a[3]);
    o.z = packbf(a[4], a[5]); o.w = packbf(a[6], a[7]);
    return o;
}
__device__ __forceinline__ v8f unpack_acc(uint4 u) {
    v8f r;
    r[0] = __uint_as_float(u.x << 16); r[1] = __uint_as_float(u.x & 0xFFFF0000u);
    r[2] = __uint_as_float(u.y << 16); r[3] = __uint_as_float(u.y & 0xFFFF0000u);
    r[4] = __uint_as_float(u.z << 16); r[5] = __uint_as_float(u.z & 0xFFFF0000u);
    r[6] = __uint_as_float(u.w << 16); r[7] = __uint_as_float(u.w & 0xFFFF0000u);
    return r;
}

// ---------- 1) prep: bf16 fragment-swizzled weights + transposed decay table ----------
__global__ void prep_kernel(const float* __restrict__ W, const float* __restrict__ U,
                            const float* __restrict__ Wd, const float* __restrict__ time_,
                            unsigned* __restrict__ ws) {
    unsigned f = blockIdx.x * blockDim.x + threadIdx.x;
    if (f < 65536u) {                       // W [256][512]
        unsigned v = f & 7u, lane = (f >> 3) & 31u, nt = (f >> 8) & 31u, kk = f >> 13;
        unsigned K = kk * 32u + frag_kin(lane, v), N = nt * 16u + (lane & 15u);
        ws[WSW_OFF + f] = packbf(W[K * GATES_ + N], W[(K + 1u) * GATES_ + N]);
    } else if (f < 98304u) {                // U [128][512]
        unsigned g = f - 65536u;
        unsigned v = g & 7u, lane = (g >> 3) & 31u, nt = (g >> 8) & 31u, kk = g >> 13;
        unsigned K = kk * 32u + frag_kin(lane, v), N = nt * 16u + (lane & 15u);
        ws[WSU_OFF + g] = packbf(U[K * GATES_ + N], U[(K + 1u) * GATES_ + N]);
    } else if (f < 106496u) {               // Wd [128][128]
        unsigned g = f - 98304u;
        unsigned v = g & 7u, lane = (g >> 3) & 31u, nt = (g >> 8) & 7u, kk = g >> 11;
        unsigned K = kk * 32u + frag_kin(lane, v), N = nt * 16u + (lane & 15u);
        ws[WSWD_OFF + g] = packbf(Wd[K * UNITS_ + N], Wd[(K + 1u) * UNITS_ + N]);
    } else if (f < 106496u + 131072u) {     // gT[t][row] = 1/log(e + time[row][t])
        unsigned g = f - 106496u;
        unsigned row = g & 127u, t = g >> 7;
        float tv = time_[(size_t)row * T_ + t];
        ((float*)(ws + GT_OFF))[g] = __fdividef(1.0f, logf(2.718281828459045f + tv));
    }
}

// ---------- 2) whole-chip GEMM: Z[b,t,:] = x[b,t,:] @ W, bf16 frags out ----------
// Written in descending t so low-t tiles are most recently resident in the
// 192 MB L2 when the scan starts reading from t = 0.
__global__ __launch_bounds__(256, 2) void gemm_xw(
    const float* __restrict__ x, const unsigned* __restrict__ wsW,
    unsigned* __restrict__ zfrag) {
    __shared__ unsigned lds_x[2048];
    const unsigned tid = threadIdx.x, wv = tid >> 5, lane = tid & 31u;
    const unsigned mt = blockIdx.x & 7u;
    const unsigned t  = (T_ - 1u) - (blockIdx.x >> 3);

    const float* xbase = x + ((size_t)(mt * 16u) * T_ + t) * D_;
    #pragma unroll
    for (unsigned j = 0; j < 8u; ++j) {     // stage x tile as bf16 A-frags
        unsigned f  = tid * 8u + j;
        unsigned v  = f & 7u, ln = (f >> 3) & 31u, kt = f >> 8;
        unsigned K  = kt * 32u + frag_kin(ln, v);
        float2 xv = *(const float2*)(xbase + (size_t)(ln & 15u) * (T_ * D_) + K);
        lds_x[f] = packbf(xv.x, xv.y);
    }
    __syncthreads();

    v8f acc0 = {}, acc1 = {}, acc2 = {}, acc3 = {};
    const unsigned nt0 = wv * 4u;
    #pragma unroll
    for (unsigned kk = 0; kk < 8u; ++kk) {
        v16bf a = *(const v16bf*)&lds_x[(kk * 32u + lane) * 8u];
        const unsigned base = kk * 8192u + lane * 8u;
        acc0 = wmma_bf16(a, *(const v16bf*)&wsW[base + (nt0 + 0u) * 256u], acc0);
        acc1 = wmma_bf16(a, *(const v16bf*)&wsW[base + (nt0 + 1u) * 256u], acc1);
        acc2 = wmma_bf16(a, *(const v16bf*)&wsW[base + (nt0 + 2u) * 256u], acc2);
        acc3 = wmma_bf16(a, *(const v16bf*)&wsW[base + (nt0 + 3u) * 256u], acc3);
    }
    size_t ob = ((((size_t)mt * T_ + t) * 32u + nt0) * 32u + lane) * 4u;
    *(uint4*)&zfrag[ob + 0u * 128u] = pack_acc(acc0);
    *(uint4*)&zfrag[ob + 1u * 128u] = pack_acc(acc1);
    *(uint4*)&zfrag[ob + 2u * 128u] = pack_acc(acc2);
    *(uint4*)&zfrag[ob + 3u * 128u] = pack_acc(acc3);
}

// ---------- 3) recurrent scan ----------
// Gate-sliced tiling: wave wv owns n-tiles {wv, 8+wv, 16+wv, 24+wv} = all four
// gates for hidden columns [16*wv, 16*wv+16). The cell update is then lane-local
// on the WMMA accumulators: the f32 cell state c lives in a persistent v8f
// fragment; only bf16 h/c A-fragments touch LDS (parity double-buffered), giving
// ONE barrier per timestep.
template <bool ZPRE>
__global__ __launch_bounds__(256, 1) void tlstm_scan(
    const float* __restrict__ x, const int* __restrict__ position,
    const float* __restrict__ b, const float* __restrict__ bd,
    const unsigned* __restrict__ ws, const unsigned* __restrict__ zfrag,
    float* __restrict__ hsel) {

    __shared__ unsigned lds_x[2][2048];   // x tile A-frags (fused mode, double buffered)
    __shared__ unsigned lds_h[2][1024];   // h bf16 A-frags [kt=4][lane][v]
    __shared__ unsigned lds_c[2][1024];   // c bf16 A-frags

    const unsigned tid  = threadIdx.x;
    const unsigned wv   = tid >> 5, lane = tid & 31u;
    const unsigned r0   = blockIdx.x * 16u;
    const unsigned hi   = lane >> 4;                 // 0: m=0..7, 1: m=8..15
    const unsigned u    = wv * 16u + (lane & 15u);   // hidden column owned by lane

    const unsigned* wsU  = ws + WSU_OFF;
    const unsigned* wsWd = ws + WSWD_OFF;
    const float*    gT   = (const float*)(ws + GT_OFF);

    // loop-invariant per-lane values
    const float bi = b[u], bf_ = b[u + 128u], bc = b[u + 256u], bo = b[u + 384u];
    const float bdv = bd[u];
    int posv[8];
    #pragma unroll
    for (unsigned v = 0; v < 8u; ++v) posv[v] = position[r0 + v + hi * 8u];
    // b16 store slot for column u in the A-fragment layout (ln varies with m)
    const unsigned kb  = u & 31u, kt = u >> 5;
    const unsigned hi2 = ((kb & 15u) >= 8u) ? 1u : 0u;
    const unsigned vv  = ((kb >> 4) & 1u) * 4u + ((kb & 7u) >> 1);
    const unsigned halfsel = u & 1u;

    for (unsigned i = tid; i < 1024u; i += 256u) { lds_h[0][i] = 0u; lds_c[0][i] = 0u; }
    if (!ZPRE) {   // stage x for t=0
        #pragma unroll
        for (unsigned j = 0; j < 8u; ++j) {
            unsigned f = tid * 8u + j;
            unsigned v = f & 7u, ln = (f >> 3) & 31u, ktx = f >> 8;
            unsigned K = ktx * 32u + frag_kin(ln, v);
            float2 xv = *(const float2*)(x + ((size_t)(r0 + (ln & 15u)) * T_) * D_ + K);
            lds_x[0][f] = packbf(xv.x, xv.y);
        }
    }
    __syncthreads();

    v8f cfrag = {};                       // persistent f32 cell state (C/D layout)

    for (int t = 0; t < T_; ++t) {
        const unsigned p = (unsigned)t & 1u;

        // ---- matrix phase: four gate tiles + cs tile for columns [16wv,16wv+16) ----
        v8f acc0, acc1, acc2, acc3, accs = {};
        if (ZPRE) {
            size_t zb = ((((size_t)blockIdx.x * T_ + (size_t)t) * 32u + wv) * 32u + lane) * 4u;
            acc0 = unpack_acc(*(const uint4*)&zfrag[zb + 0u * 1024u]);   // tile wv      (i)
            acc1 = unpack_acc(*(const uint4*)&zfrag[zb + 1u * 1024u]);   // tile wv + 8  (f)
            acc2 = unpack_acc(*(const uint4*)&zfrag[zb + 2u * 1024u]);   // tile wv + 16 (cand)
            acc3 = unpack_acc(*(const uint4*)&zfrag[zb + 3u * 1024u]);   // tile wv + 24 (o)
            if (t + 1 < T_) __builtin_prefetch(&zfrag[zb + (size_t)ZSTEP_DW], 0, 0);
        } else {
            acc0 = v8f{}; acc1 = v8f{}; acc2 = v8f{}; acc3 = v8f{};
            // launder W base each step: stream fragments from L2 instead of
            // letting the compiler hoist 256 VGPRs of them and spill.
            unsigned long long wp = (unsigned long long)(const void*)(ws + WSW_OFF);
            asm volatile("" : "+s"(wp));
            const unsigned* wsWt = (const unsigned*)wp;
            #pragma unroll
            for (unsigned kk = 0; kk < 8u; ++kk) {   // x @ W over D=256
                v16bf a = *(const v16bf*)&lds_x[p][(kk * 32u + lane) * 8u];
                const unsigned base = kk * 8192u + lane * 8u;
                acc0 = wmma_bf16(a, *(const v16bf*)&wsWt[base + (wv +  0u) * 256u], acc0);
                acc1 = wmma_bf16(a, *(const v16bf*)&wsWt[base + (wv +  8u) * 256u], acc1);
                acc2 = wmma_bf16(a, *(const v16bf*)&wsWt[base + (wv + 16u) * 256u], acc2);
                acc3 = wmma_bf16(a, *(const v16bf*)&wsWt[base + (wv + 24u) * 256u], acc3);
            }
        }
        #pragma unroll
        for (unsigned kk = 0; kk < 4u; ++kk) {       // h @ U over UNITS=128
            v16bf a = *(const v16bf*)&lds_h[p][(kk * 32u + lane) * 8u];
            const unsigned base = kk * 8192u + lane * 8u;
            acc0 = wmma_bf16(a, *(const v16bf*)&wsU[base + (wv +  0u) * 256u], acc0);
            acc1 = wmma_bf16(a, *(const v16bf*)&wsU[base + (wv +  8u) * 256u], acc1);
            acc2 = wmma_bf16(a, *(const v16bf*)&wsU[base + (wv + 16u) * 256u], acc2);
            acc3 = wmma_bf16(a, *(const v16bf*)&wsU[base + (wv + 24u) * 256u], acc3);
        }
        #pragma unroll
        for (unsigned kk = 0; kk < 4u; ++kk) {       // c @ Wd (tile wv)
            v16bf a  = *(const v16bf*)&lds_c[p][(kk * 32u + lane) * 8u];
            v16bf bb = *(const v16bf*)&wsWd[kk * 2048u + wv * 256u + lane * 8u];
            accs = wmma_bf16(a, bb, accs);
        }

        if (!ZPRE && t + 1 < T_) {   // stage next x tile into the other buffer
            const float* xbase = x + ((size_t)r0 * T_ + (size_t)(t + 1)) * D_;
            #pragma unroll
            for (unsigned j = 0; j < 8u; ++j) {
                unsigned f = tid * 8u + j;
                unsigned v = f & 7u, ln = (f >> 3) & 31u, ktx = f >> 8;
                unsigned K = ktx * 32u + frag_kin(ln, v);
                float2 xv = *(const float2*)(xbase + (size_t)(ln & 15u) * (T_ * D_) + K);
                lds_x[p ^ 1u][f] = packbf(xv.x, xv.y);
            }
        }

        // ---- lane-local cell update on the accumulators (f32 exact) ----
        {
            const float4 ga = *(const float4*)&gT[(size_t)t * B_ + r0 + hi * 8u];
            const float4 gb = *(const float4*)&gT[(size_t)t * B_ + r0 + hi * 8u + 4u];
            const float gv[8] = { ga.x, ga.y, ga.z, ga.w, gb.x, gb.y, gb.z, gb.w };
            unsigned short* hs = (unsigned short*)&lds_h[p ^ 1u][0];
            unsigned short* cs = (unsigned short*)&lds_c[p ^ 1u][0];
            #pragma unroll
            for (unsigned v = 0; v < 8u; ++v) {
                const unsigned m = v + hi * 8u;
                float csv  = tanh_fast(accs[v] + bdv);
                float cadj = cfrag[v] - csv + csv * gv[v];
                float iv = sigm(acc0[v] + bi);
                float fv = sigm(acc1[v] + bf_);
                float cd = tanh_fast(acc2[v] + bc);
                float ov = sigm(acc3[v] + bo);
                float cnw = fv * cadj + iv * cd;
                float hnw = ov * tanh_fast(cnw);
                cfrag[v] = cnw;
                if (t == posv[v]) hsel[(size_t)(r0 + m) * UNITS_ + u] = hnw;
                const unsigned ln2  = m + hi2 * 16u;
                const unsigned slot = (kt * 256u + ln2 * 8u + vv) * 2u + halfsel;
                hs[slot] = (unsigned short)f2bf1(hnw);
                cs[slot] = (unsigned short)f2bf1(cnw);
            }
        }
        __syncthreads();   // single barrier per step (parity buffers handle WAR)
    }
}

// ---------- 4) head: Dense(64)->gelu(exact)->Dense(8)->sigmoid ----------
__global__ void head_kernel(const float* __restrict__ hsel,
                            const float* __restrict__ W1, const float* __restrict__ b1,
                            const float* __restrict__ W2, const float* __restrict__ b2,
                            float* __restrict__ out) {
    int row = blockIdx.x * blockDim.x + threadIdx.x;
    if (row >= B_) return;
    const float* h = hsel + (size_t)row * UNITS_;
    float y[HID_];
    #pragma unroll 4
    for (int j = 0; j < HID_; ++j) {
        float acc = b1[j];
        for (int k = 0; k < UNITS_; ++k) acc += h[k] * W1[k * HID_ + j];
        y[j] = 0.5f * acc * (1.0f + erff(acc * 0.7071067811865476f));
    }
    #pragma unroll
    for (int o = 0; o < OUT_; ++o) {
        float acc = b2[o];
        for (int j = 0; j < HID_; ++j) acc += y[j] * W2[j * OUT_ + o];
        out[row * OUT_ + o] = 1.0f / (1.0f + __expf(-acc));
    }
}

extern "C" void kernel_launch(void* const* d_in, const int* in_sizes, int n_in,
                              void* d_out, int out_size, void* d_ws, size_t ws_size,
                              hipStream_t stream) {
    const float* x        = (const float*)d_in[0];
    const float* time_    = (const float*)d_in[1];
    const int*   position = (const int*)  d_in[2];
    const float* W        = (const float*)d_in[3];
    const float* U        = (const float*)d_in[4];
    const float* b        = (const float*)d_in[5];
    const float* Wd       = (const float*)d_in[6];
    const float* bd       = (const float*)d_in[7];
    const float* W1       = (const float*)d_in[8];
    const float* b1       = (const float*)d_in[9];
    const float* W2       = (const float*)d_in[10];
    const float* b2       = (const float*)d_in[11];

    unsigned* ws   = (unsigned*)d_ws;
    float*    hsel = (float*)(ws + HSEL_OFF);
    float*    out  = (float*)d_out;

    // 1) bf16 fragment-swizzled weights + transposed 1/log(e+t) table (~1 MB)
    prep_kernel<<<928, 256, 0, stream>>>(W, U, Wd, time_, ws);

    const size_t need_zpre = ((size_t)ZFRAG_OFF + (size_t)ZFRAG_DW) * 4ull;
    if (ws_size >= need_zpre) {
        // 2a) whole-chip x@W GEMM into bf16 fragments (~130 MB workspace)
        unsigned* zf = ws + ZFRAG_OFF;
        gemm_xw<<<8192, 256, 0, stream>>>(x, ws + WSW_OFF, zf);
        // 2b) short-critical-path scan: 20 WMMA/step/wave, 1 barrier/step
        tlstm_scan<true><<<8, 256, 0, stream>>>(x, position, b, bd, ws, zf, hsel);
    } else {
        // 2) fused fallback (52 WMMA/step/wave), W streamed from L2
        tlstm_scan<false><<<8, 256, 0, stream>>>(x, position, b, bd, ws, nullptr, hsel);
    }

    // 3) tiny head on the gathered hidden states
    head_kernel<<<1, 128, 0, stream>>>(hsel, W1, b1, W2, b2, out);
}
